// ContextAttention_42709154791805
// MI455X (gfx1250) — matrix operand order repaired and checked by reading, not compiled, not run on hardware
//
#include <hip/hip_runtime.h>
#include <hip/hip_bf16.h>
#include <math.h>

typedef __attribute__((ext_vector_type(8)))  float  v8f;
typedef __attribute__((ext_vector_type(4)))  int    v4i;
typedef __attribute__((ext_vector_type(4)))  __bf16 v4bf;
typedef __attribute__((ext_vector_type(8)))  __bf16 v8bf;
typedef __attribute__((ext_vector_type(16))) __bf16 v16bf;

#define B_  16
#define T_  4096
#define C_  1024
#define H_  16
#define HS_ 64
#define MT  32     // rows of x per block
#define XSB 1032   // LDS row stride in bf16 elems (2064 B): lane m -> bank group 4m, conflict-free

// ---------------------------------------------------------------------------
// CDNA5 async Global->LDS copy (ASYNCcnt path).
// Probed signature: (v4i addrspace(1)* src, v4i addrspace(3)* dst, imm, imm)
// ---------------------------------------------------------------------------
#if __has_builtin(__builtin_amdgcn_global_load_async_to_lds_b128)
#define HAVE_ASYNC_LDS 1
__device__ __forceinline__ void async_copy_b128(const void* g, void* l) {
  __builtin_amdgcn_global_load_async_to_lds_b128(
      (__attribute__((address_space(1))) v4i*)const_cast<void*>(g),
      (__attribute__((address_space(3))) v4i*)l, 0, 0);
}
__device__ __forceinline__ void wait_async0() {
#if __has_builtin(__builtin_amdgcn_s_wait_asynccnt)
  __builtin_amdgcn_s_wait_asynccnt(0);
#else
  asm volatile("s_wait_asynccnt 0" ::: "memory");
#endif
}
#else
#define HAVE_ASYNC_LDS 0
#endif

// Branchless tanh; prefer the CDNA5 V_TANH_F32 transcendental if exposed.
__device__ __forceinline__ float fast_tanhf(float v) {
#if __has_builtin(__builtin_amdgcn_tanhf)
  return __builtin_amdgcn_tanhf(v);
#elif __has_builtin(__builtin_amdgcn_tanh_f32)
  return __builtin_amdgcn_tanh_f32(v);
#else
  const float ax = __builtin_fabsf(v);
  const float e  = __expf(2.0f * ax);          // inf for big ax -> r == 1
  const float r  = 1.0f - 2.0f / (e + 1.0f);
  return __builtin_copysignf(r, v);
#endif
}

// ---------------------------------------------------------------------------
// Kernel 0: pre-swizzle W into bf16 hi/lo B-fragments, tile = (K=32) x (N=16).
// Fragment order: frag[(ntile*32 + ktile)*512 + lane*16 + p] where
//   n = ntile*16 + (lane&15),  k = ktile*32 + (lane>>4)*16 + p
// so each lane's 16 bf16 (32 B) are contiguous -> two coalesced b128 loads.
// ---------------------------------------------------------------------------
__global__ __launch_bounds__(256) void wprep_kernel(const float* __restrict__ W,
                                                    __bf16* __restrict__ Bh,
                                                    __bf16* __restrict__ Bl) {
  const int gid   = blockIdx.x * 256 + threadIdx.x;   // 0 .. 2^20-1
  const int p     = gid & 15;
  const int lane  = (gid >> 4) & 31;
  const int ktile = (gid >> 9) & 31;
  const int ntile = gid >> 14;
  const int n = ntile * 16 + (lane & 15);
  const int k = ktile * 32 + (lane >> 4) * 16 + p;
  const float v  = W[(size_t)n * C_ + k];
  const __bf16 hi = (__bf16)v;
  Bh[gid] = hi;
  Bl[gid] = (__bf16)(v - (float)hi);
}

// ---------------------------------------------------------------------------
// Kernel 1: fused sim[b,h,t] = sum_d tanh(x@W^T + bias) * cv  (+ masking)
// Block = 32 rows of x, 256 threads = 8 waves.
// Wave w: M-subtile = w>>2, heads (w&3)*4 .. +3.  Per head: 4 N-tiles of 16,
// K loop in steps of 32 with 3-product bf16-split WMMA (hi*hi + hi*lo + lo*hi).
// ---------------------------------------------------------------------------
__global__ __launch_bounds__(256) void sim_kernel(const float* __restrict__ x,
                                                  const int* __restrict__ mask,
                                                  const __bf16* __restrict__ Bh,
                                                  const __bf16* __restrict__ Bl,
                                                  const float* __restrict__ bias,
                                                  const float* __restrict__ cv,
                                                  float* __restrict__ sim) {
  __shared__ __bf16 xhi[MT * XSB];   // 66 KB
  __shared__ __bf16 xlo[MT * XSB];   // 66 KB
  __shared__ float  rowsum[MT];
  const int tid = threadIdx.x;
  const int blk = blockIdx.x;
  const int b  = blk >> 7;             // T_/MT = 128 tiles per batch
  const int t0 = (blk & 127) * MT;

  if (tid < MT) rowsum[tid] = 0.0f;
  __syncthreads();

  // Stage x tile as bf16 hi/lo; fold row sums (zero-padding test) into the load.
  {
    const int r = tid >> 3;   // 0..31
    const int q = tid & 7;    // 8 threads per row
    const float4* src = (const float4*)(x + ((size_t)b * T_ + (t0 + r)) * C_);
    __bf16* dh = xhi + r * XSB;
    __bf16* dl = xlo + r * XSB;
    float s = 0.0f;
#pragma unroll 4
    for (int i = q; i < C_ / 4; i += 8) {
      const float4 v = src[i];
      s += v.x + v.y + v.z + v.w;
      v4bf hv, lv;
      hv.x = (__bf16)v.x; lv.x = (__bf16)(v.x - (float)hv.x);
      hv.y = (__bf16)v.y; lv.y = (__bf16)(v.y - (float)hv.y);
      hv.z = (__bf16)v.z; lv.z = (__bf16)(v.z - (float)hv.z);
      hv.w = (__bf16)v.w; lv.w = (__bf16)(v.w - (float)hv.w);
      *(v4bf*)(dh + i * 4) = hv;       // ds_store_b64
      *(v4bf*)(dl + i * 4) = lv;
    }
    atomicAdd(&rowsum[r], s);          // ds_add_f32
  }
  __syncthreads();

  const int wave  = tid >> 5;          // 0..7
  const int msub  = wave >> 2;         // M subtile 0/1
  const int hbase = (wave & 3) * 4;    // first of 4 heads
  const int lane  = tid & 31;
  const int lm = lane & 15;            // M row (A) / N col (B,C,D)
  const int lh = lane >> 4;            // K half (A,B) / M+8 half (C,D)
  const __bf16* arow_h = xhi + (msub * 16 + lm) * XSB + lh * 16;
  const __bf16* arow_l = xlo + (msub * 16 + lm) * XSB + lh * 16;

  for (int g = 0; g < 4; ++g) {
    const int h = hbase + g;
    v8f acc[4];
#pragma unroll
    for (int j = 0; j < 4; ++j) acc[j] = 0.0f;

    for (int k0 = 0; k0 < C_; k0 += 32) {
      // A fragments: lane holds K = lh*16..+15 (contiguous, matches B fragments)
      const v8bf ah0 = *(const v8bf*)(arow_h + k0);
      const v8bf ah1 = *(const v8bf*)(arow_h + k0 + 8);
      const v8bf al0 = *(const v8bf*)(arow_l + k0);
      const v8bf al1 = *(const v8bf*)(arow_l + k0 + 8);
      const v16bf Ah = __builtin_shufflevector(ah0, ah1, 0,1,2,3,4,5,6,7,8,9,10,11,12,13,14,15);
      const v16bf Al = __builtin_shufflevector(al0, al1, 0,1,2,3,4,5,6,7,8,9,10,11,12,13,14,15);

      const int ktile = k0 >> 5;
#pragma unroll
      for (int j = 0; j < 4; ++j) {
        const size_t fidx = ((size_t)((h * 4 + j) * 32 + ktile) << 9) + lane * 16;
        __builtin_prefetch(Bh + fidx + 2048, 0, 1);   // 4 K-tiles ahead
        const v16bf Bfh = *(const v16bf*)(Bh + fidx); // 2x global_load_b128, coalesced
        const v16bf Bfl = *(const v16bf*)(Bl + fidx);
        acc[j] = __builtin_amdgcn_wmma_f32_16x16x32_bf16(
            false, Ah, false, Bfh, (short)0, acc[j], false, false);
        acc[j] = __builtin_amdgcn_wmma_f32_16x16x32_bf16(
            false, Ah, false, Bfl, (short)0, acc[j], false, false);
        acc[j] = __builtin_amdgcn_wmma_f32_16x16x32_bf16(
            false, Al, false, Bfh, (short)0, acc[j], false, false);
      }
    }

    // att = tanh(acc + bias); partial sim = att . cv  (C/D: vgpr r -> row r+8*lh)
    float p[8];
#pragma unroll
    for (int r = 0; r < 8; ++r) p[r] = 0.0f;
#pragma unroll
    for (int j = 0; j < 4; ++j) {
      const int nn = j * 16 + lm;                 // column within head
      const float bb = bias[h * HS_ + nn];
      const float cw = cv[h * HS_ + nn];
#pragma unroll
      for (int r = 0; r < 8; ++r)
        p[r] += fast_tanhf(acc[j][r] + bb) * cw;
    }
    // reduce over 16 lanes of each half-wave (masks 1,2,4,8 stay within halves)
#pragma unroll
    for (int m = 1; m < 16; m <<= 1)
#pragma unroll
      for (int r = 0; r < 8; ++r)
        p[r] += __shfl_xor(p[r], m, 32);

    if (lm == 0) {
#pragma unroll
      for (int r = 0; r < 8; ++r) {
        const int mrow = msub * 16 + r + 8 * lh;  // row within 32-row tile
        const int t = t0 + mrow;
        float v = p[r];
        if (rowsum[mrow] == 0.0f || mask[b * T_ + t] == 0) v = -1e10f;
        sim[((size_t)b * H_ + h) * T_ + t] = v;   // [B,H,T] layout
      }
    }
  }
}

// ---------------------------------------------------------------------------
// Kernel 2: softmax over T for each of the B*H = 256 rows.
// Row (16 KB) is async-staged into LDS once -> 1 global read + 1 global write.
// ---------------------------------------------------------------------------
__global__ __launch_bounds__(256) void softmax_kernel(float* __restrict__ sim) {
  __shared__ float buf[T_];    // 16 KB
  __shared__ float red[256];
  float* p = sim + (size_t)blockIdx.x * T_;
  const int tid = threadIdx.x;

#if HAVE_ASYNC_LDS
#pragma unroll
  for (int i = 0; i < 4; ++i) {
    const int c4 = i * 256 + tid;              // float4 chunk, coalesced per wave
    async_copy_b128(p + c4 * 4, buf + c4 * 4); // global_load_async_to_lds_b128
  }
  wait_async0();
  __syncthreads();
#else
  for (int t = tid; t < T_; t += 256) buf[t] = p[t];
  __syncthreads();
#endif

  float mx = -3.4e38f;
  for (int t = tid; t < T_; t += 256) mx = fmaxf(mx, buf[t]);
  red[tid] = mx;
  __syncthreads();
  for (int s = 128; s > 0; s >>= 1) {
    if (tid < s) red[tid] = fmaxf(red[tid], red[tid + s]);
    __syncthreads();
  }
  mx = red[0];
  __syncthreads();

  float sum = 0.0f;
  for (int t = tid; t < T_; t += 256) {
    const float e = __expf(buf[t] - mx);
    buf[t] = e;
    sum += e;
  }
  red[tid] = sum;
  __syncthreads();
  for (int s = 128; s > 0; s >>= 1) {
    if (tid < s) red[tid] += red[tid + s];
    __syncthreads();
  }
  const float inv = 1.0f / red[0];
  for (int t = tid; t < T_; t += 256) p[t] = buf[t] * inv;
}

// ---------------------------------------------------------------------------
// Kernel 3: pooled[b,h,d] = sum_t x[b,t,h*64+d] * w[b,h,t]   (bandwidth bound)
// The 16 KB weight row is async-staged into LDS (it is re-read by 64 lanes/elem).
// ---------------------------------------------------------------------------
__global__ __launch_bounds__(256) void pool_kernel(const float* __restrict__ x,
                                                   const float* __restrict__ w,
                                                   float* __restrict__ out) {
  __shared__ float wbuf[T_];   // 16 KB
  __shared__ float red[256];
  const int row = blockIdx.x;      // b*H + h
  const int b = row >> 4, h = row & 15;
  const int tid = threadIdx.x;
  const float* wp = w + (size_t)row * T_;

#if HAVE_ASYNC_LDS
#pragma unroll
  for (int i = 0; i < 4; ++i) {
    const int c4 = i * 256 + tid;
    async_copy_b128(wp + c4 * 4, wbuf + c4 * 4);
  }
  wait_async0();
  __syncthreads();
#else
  for (int t = tid; t < T_; t += 256) wbuf[t] = wp[t];
  __syncthreads();
#endif

  const int d = tid & 63;
  const int c = tid >> 6;          // 4-way split of T
  const float* xp = x + (size_t)b * T_ * C_ + h * HS_ + d;

  float acc = 0.0f;
  const int tb = c * (T_ / 4), te = tb + (T_ / 4);
  for (int t = tb; t < te; ++t)
    acc = fmaf(xp[(size_t)t * C_], wbuf[t], acc);

  red[tid] = acc;
  __syncthreads();
  if (tid < 64)
    out[(size_t)b * C_ + h * HS_ + d] = red[d] + red[64 + d] + red[128 + d] + red[192 + d];
}

// ---------------------------------------------------------------------------
extern "C" void kernel_launch(void* const* d_in, const int* in_sizes, int n_in,
                              void* d_out, int out_size, void* d_ws, size_t ws_size,
                              hipStream_t stream) {
  (void)in_sizes; (void)n_in; (void)out_size; (void)ws_size;
  const float* x    = (const float*)d_in[0];
  const int*   mask = (const int*)d_in[1];
  const float* W    = (const float*)d_in[2];
  const float* bias = (const float*)d_in[3];
  const float* cv   = (const float*)d_in[4];
  float* out = (float*)d_out;

  char* ws = (char*)d_ws;
  __bf16* Bh  = (__bf16*)ws;                          // 2^20 bf16 = 2 MB
  __bf16* Bl  = (__bf16*)(ws + (2u << 20));           // 2 MB
  float*  sim = (float*)(ws + (4u << 20));            // B*H*T floats = 4 MB

  wprep_kernel<<<(C_ * C_) / 256, 256, 0, stream>>>(W, Bh, Bl);
  sim_kernel<<<B_ * (T_ / MT), 256, 0, stream>>>(x, mask, Bh, Bl, bias, cv, sim);
  softmax_kernel<<<B_ * H_, 256, 0, stream>>>(sim);
  pool_kernel<<<B_ * H_, 256, 0, stream>>>(x, sim, out);
}
